// TemporalResonanceEqProp_46084999086189
// MI455X (gfx1250) — matrix operand order repaired and checked by reading, not compiled, not run on hardware
//
#include <hip/hip_runtime.h>

// ---------------------------------------------------------------------------
// Types for CDNA5 WMMA (wave32): V_WMMA_F32_16X16X32_F16
// ---------------------------------------------------------------------------
typedef __attribute__((ext_vector_type(16))) _Float16 v16h;
typedef __attribute__((ext_vector_type(8)))  _Float16 v8h;
typedef __attribute__((ext_vector_type(8)))  float    v8f;
typedef __attribute__((ext_vector_type(4)))  int      v4i;

static_assert(sizeof(v16h) == 32, "v16h must be 32 bytes");
static_assert(sizeof(v8h)  == 16, "v8h must be 16 bytes");

#define B_DIM   65536
#define IN_DIM  256
#define HID     512
#define OUT_DIM 256
#define STEPS   30

#define AS1 __attribute__((address_space(1)))
#define AS3 __attribute__((address_space(3)))

// Async global->LDS staging (GLOBAL_LOAD_ASYNC_TO_LDS_B128, ASYNCcnt) when the
// toolchain exposes it; otherwise fall back to load+ds_store.
#if defined(__gfx1250__) && __has_builtin(__builtin_amdgcn_global_load_async_to_lds_b128)
#define USE_ASYNC_STAGE 1
#else
#define USE_ASYNC_STAGE 0
#endif

#if USE_ASYNC_STAGE
__device__ __forceinline__ void async_copy16(void* lds, const void* gmem) {
    __builtin_amdgcn_global_load_async_to_lds_b128(
        (AS1 v4i*)(gmem), (AS3 v4i*)(lds), /*offset=*/0, /*cpol=*/0);
}
__device__ __forceinline__ void wait_async0() {
#if __has_builtin(__builtin_amdgcn_s_wait_asynccnt)
    __builtin_amdgcn_s_wait_asynccnt(0);
#else
    asm volatile("s_wait_asynccnt 0x0" ::: "memory");
#endif
}
#endif

__device__ __forceinline__ v8f wmma16(v16h a, v16h b, v8f c) {
    // D = A(16x32 f16) * B(32x16 f16) + C(16x16 f32)
    return __builtin_amdgcn_wmma_f32_16x16x32_f16(
        /*neg_a=*/false, a, /*neg_b=*/false, b,
        /*c_mod=*/(short)0, c, /*reuse_a=*/false, /*reuse_b=*/false);
}

__device__ __forceinline__ float fast_tanh(float x) {
#if __has_builtin(__builtin_amdgcn_tanhf)
    return __builtin_amdgcn_tanhf(x);   // v_tanh_f32 (CDNA5 trans op)
#else
    return tanhf(x);
#endif
}

// Load an A fragment (16x32 f16) from an LDS tile stored as rows of
// `rowChunks` 16-byte chunks, with chunk index XOR-swizzled by (row & 15).
// Layout: lanes 0-15 -> M=lane, K in 32kt+0..15;
//         lanes 16-31 -> M=lane-16, K in 32kt+16..31.
__device__ __forceinline__ v16h ld_a_frag(const _Float16* sA, int rowChunks,
                                          int row0, int lane, int kt) {
    int row = row0 + (lane & 15);
    int c0  = kt * 4 + ((lane >> 4) << 1); // first 16B chunk (8 f16 = 16B)
    int sw  = row & 15;
    const v8h* base = (const v8h*)sA + row * rowChunks;
    v8h lo = base[(c0)     ^ sw];
    v8h hi = base[(c0 + 1) ^ sw];         // +16 bytes = +8 K
    return __builtin_shufflevector(lo, hi, 0,1,2,3,4,5,6,7,8,9,10,11,12,13,14,15);
}

// ---------------------------------------------------------------------------
// 1) Spectral sigma: sigma = ||W v||^2 / (||W v|| + eps),  v = W^T u / (||W^T u||+eps)
// ---------------------------------------------------------------------------
__global__ __launch_bounds__(256) void sigma_kernel(
    const float* __restrict__ layers_w, const float* __restrict__ layers_u,
    const float* __restrict__ osc_w,    const float* __restrict__ osc_u,
    float* __restrict__ inv_sigma)
{
    const int l = blockIdx.x;                       // 0..3 (3 = osc)
    const float* W = (l < 3) ? (layers_w + l * HID * HID) : osc_w;
    const float* u = (l < 3) ? (layers_u + l * HID)       : osc_u;
    __shared__ float vec[HID];
    __shared__ float red[256];
    const int t = threadIdx.x;

    vec[t] = u[t]; vec[t + 256] = u[t + 256];
    __syncthreads();

    float a0 = 0.f, a1 = 0.f;                       // (W^T u)[t], [t+256]
    for (int i = 0; i < HID; ++i) {
        float ui = vec[i];
        a0 = fmaf(W[i * HID + t],       ui, a0);
        a1 = fmaf(W[i * HID + t + 256], ui, a1);
    }
    red[t] = a0 * a0 + a1 * a1;
    __syncthreads();
    for (int s = 128; s > 0; s >>= 1) { if (t < s) red[t] += red[t + s]; __syncthreads(); }
    float nrm = sqrtf(red[0]) + 1e-12f;
    __syncthreads();
    vec[t] = a0 / nrm; vec[t + 256] = a1 / nrm;     // v
    __syncthreads();

    float b0 = 0.f, b1 = 0.f;                       // (W v)[t], [t+256]
    for (int j = 0; j < HID; ++j) {
        float vj = vec[j];
        b0 = fmaf(W[t * HID + j],         vj, b0);
        b1 = fmaf(W[(t + 256) * HID + j], vj, b1);
    }
    red[t] = b0 * b0 + b1 * b1;
    __syncthreads();
    for (int s = 128; s > 0; s >>= 1) { if (t < s) red[t] += red[t + s]; __syncthreads(); }
    if (t == 0) {
        float s2 = red[0];
        float sigma = s2 / (sqrtf(s2) + 1e-12f);
        inv_sigma[l] = 1.0f / sigma;
    }
}

// ---------------------------------------------------------------------------
// 2) Weight packing into WMMA-B fragment order.
// Block(kt,nt) = 32 lanes x 16 f16 (32B/lane), linear block index given below.
// lane<16: N=nt*16+lane,   K = 32kt + 2d, 2d+1   (dword d = 0..7)
// lane>=16:N=nt*16+lane-16,K = 32kt+16+2d, 2d+1
// ---------------------------------------------------------------------------
__global__ __launch_bounds__(256) void pack_wcat_kernel(
    const float* __restrict__ layers_w, const float* __restrict__ osc_w,
    const float* __restrict__ inv_sigma, _Float16* __restrict__ wpack)
{
    int id = blockIdx.x * 256 + threadIdx.x;        // 32kt*32nt*32lane*8dw = 262144
    int dw   = id & 7;
    int lane = (id >> 3) & 31;
    int nt   = (id >> 8) & 31;
    int kt   = id >> 13;                            // 0..31 (K = 1024)
    int n  = nt * 16 + (lane & 15);
    int k0 = kt * 32 + ((lane >> 4) << 4) + dw * 2;
    float v0, v1;
    if (k0 < HID) {                                 // t-path: sum_l Wsn[l][n][k]
        float s0 = inv_sigma[0], s1 = inv_sigma[1], s2 = inv_sigma[2];
        int base = n * HID + k0;
        v0 = s0 * layers_w[base]     + s1 * layers_w[262144 + base]     + s2 * layers_w[524288 + base];
        v1 = s0 * layers_w[base + 1] + s1 * layers_w[262144 + base + 1] + s2 * layers_w[524288 + base + 1];
    } else {                                        // h-path: osc_sn[n][k-512]
        float s3 = inv_sigma[3];
        int base = n * HID + (k0 - HID);
        v0 = s3 * osc_w[base];
        v1 = s3 * osc_w[base + 1];
    }
    wpack[id * 2]     = (_Float16)v0;
    wpack[id * 2 + 1] = (_Float16)v1;
}

__global__ __launch_bounds__(256) void pack_win_kernel(
    const float* __restrict__ W_in_w, _Float16* __restrict__ winpack)
{
    int id = blockIdx.x * 256 + threadIdx.x;        // 8kt*32nt*32*8 = 65536
    int dw = id & 7, lane = (id >> 3) & 31, nt = (id >> 8) & 31, kt = id >> 13; // kt 0..7
    int n  = nt * 16 + (lane & 15);
    int k0 = kt * 32 + ((lane >> 4) << 4) + dw * 2;
    winpack[id * 2]     = (_Float16)W_in_w[n * IN_DIM + k0];
    winpack[id * 2 + 1] = (_Float16)W_in_w[n * IN_DIM + k0 + 1];
}

__global__ __launch_bounds__(256) void pack_head_kernel(
    const float* __restrict__ head_w, _Float16* __restrict__ headpack)
{
    int id = blockIdx.x * 256 + threadIdx.x;        // 16kt*16nt*32*8 = 65536
    int dw = id & 7, lane = (id >> 3) & 31, nt = (id >> 8) & 15, kt = id >> 12; // kt 0..15
    int n  = nt * 16 + (lane & 15);
    int k0 = kt * 32 + ((lane >> 4) << 4) + dw * 2;
    headpack[id * 2]     = (_Float16)head_w[n * HID + k0];
    headpack[id * 2 + 1] = (_Float16)head_w[n * HID + k0 + 1];
}

// ---------------------------------------------------------------------------
// 3) Embed: xemb = x @ W_in^T + W_in_b + sum_l layers_b ; zero-init state.
//    Block: 32 rows x 512 cols (8 waves x N64). K = 256.
//    (Manual staging: converts f32 -> f16 in flight.)
// ---------------------------------------------------------------------------
__global__ __launch_bounds__(256) void embed_kernel(
    const float* __restrict__ x, const _Float16* __restrict__ winpack,
    const float* __restrict__ W_in_b, const float* __restrict__ layers_b,
    float* __restrict__ xemb, float* __restrict__ h32, _Float16* __restrict__ abuf)
{
    __shared__ _Float16 sA[32 * IN_DIM];            // 16 KB, 32 chunks/row
    const int t = threadIdx.x, lane = t & 31, wid = t >> 5;
    const int m_base = blockIdx.x * 32;

    #pragma unroll
    for (int i = 0; i < 4; ++i) {                   // stage + f32->f16 convert
        int lin = t + i * 256;
        int row = lin >> 5, c = lin & 31;
        const float* g = x + (size_t)(m_base + row) * IN_DIM + c * 8;
        v8h hv;
        #pragma unroll
        for (int j = 0; j < 8; ++j) hv[j] = (_Float16)g[j];
        ((v8h*)sA)[row * 32 + (c ^ (row & 15))] = hv;
    }
    __syncthreads();

    const int nt0 = wid * 4;
    v8f acc[2][4];
    #pragma unroll
    for (int h = 0; h < 2; ++h)
        #pragma unroll
        for (int s = 0; s < 4; ++s) acc[h][s] = (v8f)(0.0f);

    for (int kt = 0; kt < 8; ++kt) {
        v16h a0 = ld_a_frag(sA, 32, 0,  lane, kt);
        v16h a1 = ld_a_frag(sA, 32, 16, lane, kt);
        #pragma unroll
        for (int s = 0; s < 4; ++s) {
            v16h b = *(const v16h*)(winpack + (size_t)(kt * 32 + nt0 + s) * 512 + lane * 16);
            acc[0][s] = wmma16(a0, b, acc[0][s]);
            acc[1][s] = wmma16(a1, b, acc[1][s]);
        }
    }

    const int ncol = lane & 15, mo = (lane >> 4) << 3;
    #pragma unroll
    for (int h = 0; h < 2; ++h)
        #pragma unroll
        for (int s = 0; s < 4; ++s) {
            int n = wid * 64 + s * 16 + ncol;
            float bias = W_in_b[n] + layers_b[n] + layers_b[HID + n] + layers_b[2 * HID + n];
            #pragma unroll
            for (int r = 0; r < 8; ++r) {
                int m   = m_base + h * 16 + mo + r;
                int idx = m * HID + n;
                xemb[idx] = acc[h][s][r] + bias;
                h32[idx]  = 0.0f;                   // h0 = 0
                abuf[m * 1024 + n]       = (_Float16)0.0f;   // t16
                abuf[m * 1024 + HID + n] = (_Float16)0.0f;   // h16
            }
        }
}

// ---------------------------------------------------------------------------
// 4) Fused step: acc = [t16|h16] @ Wcat (K=1024) ; epilogue tanh blend.
//    Block: 32 rows x 512 cols, 8 waves, wave tile 32x64, 256 WMMAs/wave.
// ---------------------------------------------------------------------------
__global__ __launch_bounds__(256) void step_kernel(
    _Float16* __restrict__ abuf, const _Float16* __restrict__ wpack,
    const float* __restrict__ xemb, float* __restrict__ h32)
{
    __shared__ _Float16 sA[32 * 1024];              // 64 KB, 128 chunks/row
    const int t = threadIdx.x, lane = t & 31, wid = t >> 5;
    const int m_base = blockIdx.x * 32;

    {   // stage A tile (rows m_base..m_base+31 of [t16|h16]) with XOR swizzle
#if USE_ASYNC_STAGE
        const char* gA = (const char*)(abuf + (size_t)m_base * 1024);
        char* sc = (char*)sA;
        #pragma unroll
        for (int i = 0; i < 16; ++i) {
            int lin = t + i * 256;
            int row = lin >> 7, c = lin & 127;
            async_copy16(sc + (size_t)(row * 128 + (c ^ (row & 15))) * 16,
                         gA + (size_t)(row * 128 + c) * 16);
        }
        wait_async0();
#else
        const v8h* gA = (const v8h*)(abuf + (size_t)m_base * 1024);
        v8h* sc = (v8h*)sA;
        #pragma unroll
        for (int i = 0; i < 16; ++i) {
            int lin = t + i * 256;
            int row = lin >> 7, c = lin & 127;
            sc[row * 128 + (c ^ (row & 15))] = gA[row * 128 + c];
        }
#endif
    }
    __syncthreads();

    const int nt0 = wid * 4;
    v8f acc[2][4];
    #pragma unroll
    for (int h = 0; h < 2; ++h)
        #pragma unroll
        for (int s = 0; s < 4; ++s) acc[h][s] = (v8f)(0.0f);

    for (int kt = 0; kt < 32; ++kt) {
        v16h a0 = ld_a_frag(sA, 128, 0,  lane, kt);
        v16h a1 = ld_a_frag(sA, 128, 16, lane, kt);
        #pragma unroll
        for (int s = 0; s < 4; ++s) {
            v16h b = *(const v16h*)(wpack + (size_t)(kt * 32 + nt0 + s) * 512 + lane * 16);
            acc[0][s] = wmma16(a0, b, acc[0][s]);
            acc[1][s] = wmma16(a1, b, acc[1][s]);
        }
    }

    const int ncol = lane & 15, mo = (lane >> 4) << 3;
    #pragma unroll
    for (int h = 0; h < 2; ++h)
        #pragma unroll
        for (int s = 0; s < 4; ++s) {
            int n = wid * 64 + s * 16 + ncol;
            #pragma unroll
            for (int r = 0; r < 8; ++r) {
                int m   = m_base + h * 16 + mo + r;
                int idx = m * HID + n;
                float pre = acc[h][s][r] + xemb[idx];   // xemb includes all biases
                float htg = fast_tanh(pre);
                float hn  = 0.5f * h32[idx] + 0.5f * htg;   // ALPHA = 0.5
                h32[idx] = hn;
                abuf[m * 1024 + n]       = (_Float16)fast_tanh(hn);  // next t16
                abuf[m * 1024 + HID + n] = (_Float16)hn;             // next h16
            }
        }
}

// ---------------------------------------------------------------------------
// 5) Head: out = h @ head_w^T + head_b.  Block: 32 rows x 256 cols, 4 waves.
// ---------------------------------------------------------------------------
__global__ __launch_bounds__(128) void head_kernel(
    const _Float16* __restrict__ abuf, const _Float16* __restrict__ headpack,
    const float* __restrict__ head_b, float* __restrict__ out)
{
    __shared__ _Float16 sA[32 * HID];               // 32 KB, 64 chunks/row
    const int t = threadIdx.x, lane = t & 31, wid = t >> 5;  // wid 0..3
    const int m_base = blockIdx.x * 32;

#if USE_ASYNC_STAGE
    {
        const char* gA = (const char*)abuf;
        char* sc = (char*)sA;
        #pragma unroll
        for (int i = 0; i < 16; ++i) {              // stage h16 half of abuf rows
            int lin = t + i * 128;
            int row = lin >> 6, c = lin & 63;
            async_copy16(sc + (size_t)(row * 64 + (c ^ (row & 15))) * 16,
                         gA + (size_t)(m_base + row) * 2048 + (size_t)HID * 2 + (size_t)c * 16);
        }
        wait_async0();
    }
#else
    #pragma unroll
    for (int i = 0; i < 16; ++i) {                  // stage h16 half of abuf rows
        int lin = t + i * 128;
        int row = lin >> 6, c = lin & 63;
        ((v8h*)sA)[row * 64 + (c ^ (row & 15))] =
            *(const v8h*)(abuf + (size_t)(m_base + row) * 1024 + HID + c * 8);
    }
#endif
    __syncthreads();

    const int nt0 = wid * 4;
    v8f acc[2][4];
    #pragma unroll
    for (int h = 0; h < 2; ++h)
        #pragma unroll
        for (int s = 0; s < 4; ++s) acc[h][s] = (v8f)(0.0f);

    for (int kt = 0; kt < 16; ++kt) {
        v16h a0 = ld_a_frag(sA, 64, 0,  lane, kt);
        v16h a1 = ld_a_frag(sA, 64, 16, lane, kt);
        #pragma unroll
        for (int s = 0; s < 4; ++s) {
            v16h b = *(const v16h*)(headpack + (size_t)(kt * 16 + nt0 + s) * 512 + lane * 16);
            acc[0][s] = wmma16(a0, b, acc[0][s]);
            acc[1][s] = wmma16(a1, b, acc[1][s]);
        }
    }

    const int ncol = lane & 15, mo = (lane >> 4) << 3;
    #pragma unroll
    for (int h = 0; h < 2; ++h)
        #pragma unroll
        for (int s = 0; s < 4; ++s) {
            int n = wid * 64 + s * 16 + ncol;
            float bias = head_b[n];
            #pragma unroll
            for (int r = 0; r < 8; ++r) {
                int m = m_base + h * 16 + mo + r;
                out[m * OUT_DIM + n] = acc[h][s][r] + bias;
            }
        }
}

// ---------------------------------------------------------------------------
// Launch: sigma -> packs -> embed -> 30x step -> head   (all on `stream`)
// ---------------------------------------------------------------------------
extern "C" void kernel_launch(void* const* d_in, const int* in_sizes, int n_in,
                              void* d_out, int out_size, void* d_ws, size_t ws_size,
                              hipStream_t stream) {
    (void)in_sizes; (void)n_in; (void)out_size; (void)ws_size;
    const float* x        = (const float*)d_in[0];
    const float* W_in_w   = (const float*)d_in[1];
    const float* W_in_b   = (const float*)d_in[2];
    const float* layers_w = (const float*)d_in[3];
    const float* layers_b = (const float*)d_in[4];
    const float* layers_u = (const float*)d_in[5];
    const float* osc_w    = (const float*)d_in[6];
    const float* osc_u    = (const float*)d_in[7];
    const float* head_w   = (const float*)d_in[8];
    const float* head_b   = (const float*)d_in[9];
    float* out = (float*)d_out;

    char* ws = (char*)d_ws;
    float*    h32      = (float*)(ws);                          // 128 MiB
    float*    xemb     = (float*)(ws + (128ull << 20));         // 128 MiB
    _Float16* abuf     = (_Float16*)(ws + (256ull << 20));      // 128 MiB  [t16|h16]
    _Float16* wpack    = (_Float16*)(ws + (384ull << 20));      //   1 MiB
    _Float16* winpack  = (_Float16*)(ws + (385ull << 20));      // 256 KiB
    _Float16* headpack = (_Float16*)(ws + (385ull << 20) + (512ull << 10));
    float*    invsig   = (float*)(ws + (386ull << 20));

    sigma_kernel<<<4, 256, 0, stream>>>(layers_w, layers_u, osc_w, osc_u, invsig);
    pack_wcat_kernel<<<1024, 256, 0, stream>>>(layers_w, osc_w, invsig, wpack);
    pack_win_kernel<<<256, 256, 0, stream>>>(W_in_w, winpack);
    pack_head_kernel<<<256, 256, 0, stream>>>(head_w, headpack);

    embed_kernel<<<B_DIM / 32, 256, 0, stream>>>(x, winpack, W_in_b, layers_b,
                                                 xemb, h32, abuf);
    for (int s = 0; s < STEPS; ++s)
        step_kernel<<<B_DIM / 32, 256, 0, stream>>>(abuf, wpack, xemb, h32);

    head_kernel<<<B_DIM / 32, 128, 0, stream>>>(abuf, headpack, head_b, out);
}
